// GAT_87084756893762
// MI455X (gfx1250) — compile-verified, matrix operand order checked by reading, not run on hardware
//
#include <hip/hip_runtime.h>
#include <hip/hip_bf16.h>
#include <math.h>

typedef _Float16 half_t;
typedef __attribute__((ext_vector_type(16))) _Float16 v16h;
typedef __attribute__((ext_vector_type(8)))  _Float16 v8h;
typedef __attribute__((ext_vector_type(8)))  float    v8f;

#define NEG_SLOPE 0.2f

// ---- ordered-uint encoding of float for atomicMax (monotonic) ----
__device__ __forceinline__ unsigned fkey(float f) {
  unsigned b = __float_as_uint(f);
  return (b & 0x80000000u) ? ~b : (b | 0x80000000u);
}
__device__ __forceinline__ float funkey(unsigned k) {
  return (k & 0x80000000u) ? __uint_as_float(k & 0x7FFFFFFFu)
                           : __uint_as_float(~k);
}

// ============================================================
// Pack B matrices col-major f16: B0t[c][k] (128x128), B1t[c][k] (64x64)
// ============================================================
__global__ void gat_pack_B(const float* __restrict__ W0, const float* __restrict__ sW0,
                           const float* __restrict__ W1, const float* __restrict__ sW1,
                           half_t* __restrict__ B0t, half_t* __restrict__ B1t) {
  for (int i = blockIdx.x * blockDim.x + threadIdx.x; i < 128 * 128;
       i += gridDim.x * blockDim.x) {
    int c = i >> 7, k = i & 127;
    float v = (c < 64) ? W0[k * 64 + c] : sW0[k * 64 + (c - 64)];
    B0t[c * 128 + k] = (half_t)v;
  }
  for (int i = blockIdx.x * blockDim.x + threadIdx.x; i < 64 * 64;
       i += gridDim.x * blockDim.x) {
    int c = i >> 6, k = i & 63;
    float v = (c < 32) ? W1[k * 32 + c] : sW1[k * 32 + (c - 32)];
    B1t[c * 64 + k] = (half_t)v;
  }
}

// ============================================================
// GEMM0: h0 = x@W0 ; skip0 = x@sW0 + sb0   (K=128, 128 out cols)
// 256 threads = 8 waves; each wave owns a 16x16 output tile.
// ============================================================
__global__ void __launch_bounds__(256)
gat_gemm0(const float* __restrict__ x, const half_t* __restrict__ B0t,
          const float* __restrict__ sb0, float* __restrict__ h0,
          float* __restrict__ skip0, int n_nodes) {
  __shared__ half_t As[16 * 136];   // 136-half row stride (272B, 16B aligned)
  const int row0 = blockIdx.x * 16;
  const int tid = threadIdx.x;
  const int wave = tid >> 5, lane = tid & 31;
  const int colbase = wave * 16;
  const int m = lane & 15;              // A row / B col for this lane
  const int kb = (lane >> 4) * 8;       // K sub-offset per ISA 16-bit layout

  // warm L2/L0 for this wave's B slab while we stage A (global_prefetch_b8)
  __builtin_prefetch(&B0t[(size_t)(colbase + m) * 128], 0, 3);

  // stage A tile f32 -> f16 in LDS
  for (int i = tid; i < 16 * 128; i += 256) {
    int am = i >> 7, k = i & 127;
    int r = row0 + am;
    float v = (r < n_nodes) ? x[(size_t)r * 128 + k] : 0.f;
    As[am * 136 + k] = (half_t)v;
  }
  __syncthreads();

  // epilogue routing (uniform per lane; resolved with selects, no branches)
  const int col = colbase + (lane & 15);
  const int cc = col & 63;
  const bool isSkip = col >= 64;
  float bias = sb0[cc];                       // always a valid load
  bias = isSkip ? bias : 0.f;
  float* __restrict__ base = isSkip ? skip0 : h0;

  v8f acc = {};
#pragma unroll
  for (int k0 = 0; k0 < 128; k0 += 32) {
    v8h alo = *(const v8h*)(&As[m * 136 + k0 + kb]);
    v8h ahi = *(const v8h*)(&As[m * 136 + k0 + kb + 16]);
    const half_t* bp = &B0t[(size_t)(colbase + m) * 128 + k0 + kb];
    v8h blo = *(const v8h*)(bp);
    v8h bhi = *(const v8h*)(bp + 16);
    v16h a, b;
#pragma unroll
    for (int j = 0; j < 8; ++j) {
      a[j] = alo[j]; a[8 + j] = ahi[j];
      b[j] = blo[j]; b[8 + j] = bhi[j];
    }
    acc = __builtin_amdgcn_wmma_f32_16x16x32_f16(false, a, false, b,
                                                 (short)0, acc, false, false);
  }

  // C layout: VGPR r -> row = r + 8*(lane>>4); col = lane&15
  const int mbase = row0 + (lane >> 4) * 8;
  if (row0 + 16 <= n_nodes) {
#pragma unroll
    for (int r = 0; r < 8; ++r)
      base[(size_t)(mbase + r) * 64 + cc] = acc[r] + bias;
  } else {
#pragma unroll
    for (int r = 0; r < 8; ++r) {
      int rr = mbase + r;
      if (rr < n_nodes) base[(size_t)rr * 64 + cc] = acc[r] + bias;
    }
  }
}

// ============================================================
// GEMM1: g1 = h1@W1 ; skip1 = h1@sW1 + sb1  (K=64, 64 out cols)
// 128 threads = 4 waves; A already f16 in global.
// ============================================================
__global__ void __launch_bounds__(128)
gat_gemm1(const half_t* __restrict__ h1h, const half_t* __restrict__ B1t,
          const float* __restrict__ sb1, float* __restrict__ g1,
          float* __restrict__ skip1, int n_nodes) {
  const int row0 = blockIdx.x * 16;
  const int tid = threadIdx.x;
  const int wave = tid >> 5, lane = tid & 31;
  const int colbase = wave * 16;
  const int m = lane & 15;
  const int kb = (lane >> 4) * 8;
  const int arow = min(row0 + m, n_nodes - 1);   // clamp (N%16==0 normally)

  const int col = colbase + (lane & 15);
  const int cc = col & 31;
  const bool isSkip = col >= 32;
  float bias = sb1[cc];
  bias = isSkip ? bias : 0.f;
  float* __restrict__ base = isSkip ? skip1 : g1;

  v8f acc = {};
#pragma unroll
  for (int k0 = 0; k0 < 64; k0 += 32) {
    const half_t* ap = &h1h[(size_t)arow * 64 + k0 + kb];
    v8h alo = *(const v8h*)(ap);
    v8h ahi = *(const v8h*)(ap + 16);
    const half_t* bp = &B1t[(size_t)(colbase + m) * 64 + k0 + kb];
    v8h blo = *(const v8h*)(bp);
    v8h bhi = *(const v8h*)(bp + 16);
    v16h a, b;
#pragma unroll
    for (int j = 0; j < 8; ++j) {
      a[j] = alo[j]; a[8 + j] = ahi[j];
      b[j] = blo[j]; b[8 + j] = bhi[j];
    }
    acc = __builtin_amdgcn_wmma_f32_16x16x32_f16(false, a, false, b,
                                                 (short)0, acc, false, false);
  }

  const int mbase = row0 + (lane >> 4) * 8;
  if (row0 + 16 <= n_nodes) {
#pragma unroll
    for (int r = 0; r < 8; ++r)
      base[(size_t)(mbase + r) * 32 + cc] = acc[r] + bias;
  } else {
#pragma unroll
    for (int r = 0; r < 8; ++r) {
      int rr = mbase + r;
      if (rr < n_nodes) base[(size_t)rr * 32 + cc] = acc[r] + bias;
    }
  }
}

// ============================================================
// Per-node attention scores: s_src[i] = h[i,:]·a_src ; s_dst likewise
// ============================================================
__global__ void gat_scores(const float* __restrict__ h, const float* __restrict__ a_src,
                           const float* __restrict__ a_dst, float* __restrict__ ssrc,
                           float* __restrict__ sdst, int n_nodes, int C) {
  int i = blockIdx.x * blockDim.x + threadIdx.x;
  if (i >= n_nodes) return;
  float s1 = 0.f, s2 = 0.f;
  for (int c = 0; c < C; ++c) {
    float v = h[(size_t)i * C + c];
    s1 = fmaf(v, a_src[c], s1);
    s2 = fmaf(v, a_dst[c], s2);
  }
  ssrc[i] = s1;
  sdst[i] = s2;
}

__device__ __forceinline__ void edge_sd(const int* __restrict__ ei, int e, int E_,
                                        int& s, int& d) {
  if (e < E_) { s = ei[e]; d = ei[E_ + e]; }
  else        { s = e - E_; d = e - E_; }   // self loop
}

// ---- pass A: segment max via ordered-uint atomicMax ----
__global__ void gat_edge_max(const int* __restrict__ ei, const float* __restrict__ ssrc,
                             const float* __restrict__ sdst, unsigned* __restrict__ mbits,
                             int E_, int n_nodes) {
  int e = blockIdx.x * blockDim.x + threadIdx.x;
  if (e >= E_ + n_nodes) return;
  int s, d; edge_sd(ei, e, E_, s, d);
  float v = ssrc[s] + sdst[d];
  v = (v > 0.f) ? v : NEG_SLOPE * v;
  atomicMax(&mbits[d], fkey(v));
}

// ---- pass B: p = exp(e - max); denom += p ----
__global__ void gat_edge_exp(const int* __restrict__ ei, const float* __restrict__ ssrc,
                             const float* __restrict__ sdst, const unsigned* __restrict__ mbits,
                             float* __restrict__ pbuf, float* __restrict__ denom,
                             int E_, int n_nodes) {
  int e = blockIdx.x * blockDim.x + threadIdx.x;
  if (e >= E_ + n_nodes) return;
  int s, d; edge_sd(ei, e, E_, s, d);
  float v = ssrc[s] + sdst[d];
  v = (v > 0.f) ? v : NEG_SLOPE * v;
  float p = expf(v - funkey(mbits[d]));
  pbuf[e] = p;
  atomicAdd(&denom[d], p);
}

// ---- pass C: agg[d,:] += (p/denom[d]) * h[s,:]; thread per (edge, channel) ----
__global__ void gat_edge_agg(const int* __restrict__ ei, const float* __restrict__ pbuf,
                             const float* __restrict__ denom, const float* __restrict__ h,
                             float* __restrict__ agg, int E_, int n_nodes, int logC) {
  long long idx = (long long)blockIdx.x * blockDim.x + threadIdx.x;
  const int C = 1 << logC;
  long long total = (long long)(E_ + n_nodes) << logC;
  if (idx >= total) return;
  int e = (int)(idx >> logC);
  int c = (int)(idx & (C - 1));
  int s, d; edge_sd(ei, e, E_, s, d);
  float alpha = pbuf[e] / denom[d];
  atomicAdd(&agg[((size_t)d << logC) + c], alpha * h[((size_t)s << logC) + c]);
}

// ---- finalize layer0: h1 = elu(agg0 + b0 + skip0), emitted as f16 ----
__global__ void gat_finalize0(const float* __restrict__ agg0, const float* __restrict__ b0,
                              const float* __restrict__ skip0, half_t* __restrict__ h1h,
                              int n_nodes) {
  long long idx = (long long)blockIdx.x * blockDim.x + threadIdx.x;
  if (idx >= (long long)n_nodes * 64) return;
  int c = (int)(idx & 63);
  float v = agg0[idx] + b0[c] + skip0[idx];
  v = (v > 0.f) ? v : (expf(v) - 1.f);   // ELU
  h1h[idx] = (half_t)v;
}

// ---- finalize layer1 + log_softmax: one wave32 per node (32 channels) ----
__global__ void __launch_bounds__(256)
gat_finalize1(const float* __restrict__ agg1, const float* __restrict__ b1,
              const float* __restrict__ skip1, float* __restrict__ out, int n_nodes) {
  int node = blockIdx.x * 8 + (threadIdx.x >> 5);
  int lane = threadIdx.x & 31;
  if (node >= n_nodes) return;
  size_t idx = (size_t)node * 32 + lane;
  float z = agg1[idx] + b1[lane] + skip1[idx];
  float m = z;
#pragma unroll
  for (int off = 16; off > 0; off >>= 1) m = fmaxf(m, __shfl_xor(m, off, 32));
  float ex = expf(z - m);
  float s = ex;
#pragma unroll
  for (int off = 16; off > 0; off >>= 1) s += __shfl_xor(s, off, 32);
  out[idx] = z - m - logf(s);
}

// ============================================================
extern "C" void kernel_launch(void* const* d_in, const int* in_sizes, int n_in,
                              void* d_out, int out_size, void* d_ws, size_t ws_size,
                              hipStream_t stream) {
  const float* x      = (const float*)d_in[0];
  const int*   ei     = (const int*)d_in[1];
  const float* W0     = (const float*)d_in[2];
  const float* a_src0 = (const float*)d_in[3];
  const float* a_dst0 = (const float*)d_in[4];
  const float* b0     = (const float*)d_in[5];
  const float* sW0    = (const float*)d_in[6];
  const float* sb0    = (const float*)d_in[7];
  const float* W1     = (const float*)d_in[8];
  const float* a_src1 = (const float*)d_in[9];
  const float* a_dst1 = (const float*)d_in[10];
  const float* b1     = (const float*)d_in[11];
  const float* sW1    = (const float*)d_in[12];
  const float* sb1    = (const float*)d_in[13];
  float* out = (float*)d_out;

  const int N_ = in_sizes[0] / 128;
  const int E_ = in_sizes[1] / 2;
  const int EN = E_ + N_;

  // ---- workspace layout (256B aligned); layer1 reuses layer0 regions ----
  char* ws = (char*)d_ws;
  size_t o = 0;
  auto alloc = [&](size_t bytes) { size_t r = o; o += (bytes + 255) & ~(size_t)255; return r; };
  half_t*   B0t   = (half_t*)(ws + alloc(128 * 128 * sizeof(half_t)));
  half_t*   B1t   = (half_t*)(ws + alloc(64 * 64 * sizeof(half_t)));
  float*    h0    = (float*)(ws + alloc((size_t)N_ * 64 * sizeof(float)));
  float*    skip0 = (float*)(ws + alloc((size_t)N_ * 64 * sizeof(float)));
  float*    agg0  = (float*)(ws + alloc((size_t)N_ * 64 * sizeof(float)));
  half_t*   h1h   = (half_t*)(ws + alloc((size_t)N_ * 64 * sizeof(half_t)));
  float*    ssrc  = (float*)(ws + alloc((size_t)N_ * sizeof(float)));
  float*    sdst  = (float*)(ws + alloc((size_t)N_ * sizeof(float)));
  unsigned* mbits = (unsigned*)(ws + alloc((size_t)N_ * sizeof(unsigned)));
  float*    denom = (float*)(ws + alloc((size_t)N_ * sizeof(float)));
  float*    pbuf  = (float*)(ws + alloc((size_t)EN * sizeof(float)));
  // layer1 aliases (layer0 buffers dead by then):
  float* g1    = h0;                       // N x 32
  float* skip1 = h0 + (size_t)N_ * 32;     // N x 32 (second half of h0 region)
  float* agg1  = agg0;                     // N x 32

  const int rowTiles = (N_ + 15) / 16;
  const int TB = 256;

  // 1) pack weights to f16 col-major
  gat_pack_B<<<64, TB, 0, stream>>>(W0, sW0, W1, sW1, B0t, B1t);

  // 2) layer0 GEMM (WMMA) -> h0, skip0(+sb0)
  gat_gemm0<<<rowTiles, 256, 0, stream>>>(x, B0t, sb0, h0, skip0, N_);

  // 3) attention scores layer0
  gat_scores<<<(N_ + TB - 1) / TB, TB, 0, stream>>>(h0, a_src0, a_dst0, ssrc, sdst, N_, 64);

  // 4) init segment buffers (fkey init 0 is safe: every dst has a self loop)
  hipMemsetAsync(mbits, 0, (size_t)N_ * 4, stream);
  hipMemsetAsync(denom, 0, (size_t)N_ * 4, stream);
  hipMemsetAsync(agg0, 0, (size_t)N_ * 64 * 4, stream);

  // 5-7) edge softmax + aggregation, layer0 (C=64)
  gat_edge_max<<<(EN + TB - 1) / TB, TB, 0, stream>>>(ei, ssrc, sdst, mbits, E_, N_);
  gat_edge_exp<<<(EN + TB - 1) / TB, TB, 0, stream>>>(ei, ssrc, sdst, mbits, pbuf, denom, E_, N_);
  {
    long long tot = (long long)EN << 6;
    gat_edge_agg<<<(unsigned)((tot + TB - 1) / TB), TB, 0, stream>>>(ei, pbuf, denom, h0, agg0, E_, N_, 6);
  }

  // 8) finalize layer0 (bias + skip + ELU) -> h1 in f16
  {
    long long tot = (long long)N_ * 64;
    gat_finalize0<<<(unsigned)((tot + TB - 1) / TB), TB, 0, stream>>>(agg0, b0, skip0, h1h, N_);
  }

  // 9) layer1 GEMM (WMMA) -> g1, skip1(+sb1)
  gat_gemm1<<<rowTiles, 128, 0, stream>>>(h1h, B1t, sb1, g1, skip1, N_);

  // 10) attention scores layer1
  gat_scores<<<(N_ + TB - 1) / TB, TB, 0, stream>>>(g1, a_src1, a_dst1, ssrc, sdst, N_, 32);

  // 11) re-init segment buffers
  hipMemsetAsync(mbits, 0, (size_t)N_ * 4, stream);
  hipMemsetAsync(denom, 0, (size_t)N_ * 4, stream);
  hipMemsetAsync(agg1, 0, (size_t)N_ * 32 * 4, stream);

  // 12) edge softmax + aggregation, layer1 (C=32)
  gat_edge_max<<<(EN + TB - 1) / TB, TB, 0, stream>>>(ei, ssrc, sdst, mbits, E_, N_);
  gat_edge_exp<<<(EN + TB - 1) / TB, TB, 0, stream>>>(ei, ssrc, sdst, mbits, pbuf, denom, E_, N_);
  {
    long long tot = (long long)EN << 5;
    gat_edge_agg<<<(unsigned)((tot + TB - 1) / TB), TB, 0, stream>>>(ei, pbuf, denom, g1, agg1, E_, N_, 5);
  }

  // 13) finalize layer1 + log_softmax (wave32 per node) -> out
  gat_finalize1<<<(N_ + 7) / 8, 256, 0, stream>>>(agg1, b1, skip1, out, N_);
}
